// WordNetEmbedding_61924838474211
// MI455X (gfx1250) — compile-verified, matrix operand order checked by reading, not run on hardware
//
#include <hip/hip_runtime.h>

// ---------------------------------------------------------------------------
// WordNet embedding fusion for gfx1250 (MI455X, wave32, WMMA)
//   out[b,s,h] = (mean of <=4 gathered emb rows)[b,s,:300] @ W[300,768]
//               + bias[h] + input_tensors[b,s,h]
// HBM-bound (~450 MB streamed); f16 WMMA w/ fp32 accum. Wave tiling minimizes
// L2 read amplification of the B panel (each B fragment feeds 4 WMMAs), and
// the single-touch streams (input_tensors / out) use non-temporal cache
// policy so emb_table (49 MB) and Wt stay L2-resident.
// ---------------------------------------------------------------------------

typedef __attribute__((ext_vector_type(16))) _Float16 v16h;
typedef __attribute__((ext_vector_type(8)))  float    v8f;

#define EMB      300      // embedding dim (K, logical)
#define KP       320      // padded K for Wt (multiple of 32)
#define AKP      328      // padded K for LDS A panel (stride 656B: conflict-free, 16B aligned)
#define HID      768      // hidden dim (N)
#define MTILE    64       // tokens per block (4 WMMA M-tiles)
#define NTHREADS 256      // 8 waves of 32
#define KSTEPS   (KP / 32)

union FragU {
    uint4 u[2];
    v16h  v;
};

// ---- prep: W[300][768] f32 -> Wt[768][320] f16 (transposed, zero-padded K) --
__global__ __launch_bounds__(NTHREADS)
void wt_build_kernel(const float* __restrict__ W, _Float16* __restrict__ Wt) {
    int idx = blockIdx.x * NTHREADS + threadIdx.x;   // over HID*KP
    if (idx >= HID * KP) return;
    int h = idx / KP;
    int d = idx - h * KP;
    float v = (d < EMB) ? W[d * HID + h] : 0.0f;
    Wt[idx] = (_Float16)v;
}

// ---- fused gather-average + WMMA GEMM + bias + residual ---------------------
__global__ __launch_bounds__(NTHREADS)
void wordnet_fused_kernel(const int*      __restrict__ ids,        // [B*S]
                          const float*    __restrict__ xin,        // [B*S,768]
                          const int*      __restrict__ wn_map,     // [VOCAB,4]
                          const int*      __restrict__ wn_counts,  // [VOCAB]
                          const float*    __restrict__ emb,        // [N_ENT,300]
                          const _Float16* __restrict__ Wt,         // [768,KP] f16
                          const float*    __restrict__ bias,       // [768]
                          float*          __restrict__ out)        // [B*S,768]
{
    __shared__ __align__(16) _Float16 sA[MTILE * AKP];   // 64 x 328 f16 = 41 KB
    __shared__ int   sRows[MTILE][4];
    __shared__ float sInv[MTILE];
    __shared__ int   sCnt[MTILE];

    const int tid  = threadIdx.x;
    const int tok0 = blockIdx.x * MTILE;

    // ---- token metadata into LDS -------------------------------------------
    if (tid < MTILE) {
        int id  = ids[tok0 + tid];
        int cnt = wn_counts[id];
        sCnt[tid] = cnt;
        sInv[tid] = 1.0f / (float)(cnt > 0 ? cnt : 1);
        sRows[tid][0] = wn_map[id * 4 + 0];
        sRows[tid][1] = wn_map[id * 4 + 1];
        sRows[tid][2] = wn_map[id * 4 + 2];
        sRows[tid][3] = wn_map[id * 4 + 3];
    }
    __syncthreads();

    // ---- ragged gather + average -> f16 A panel in LDS ---------------------
    // 64*328 = 20992 = 82*256 -> perfectly uniform loop (EXEC stays full).
    // emb_table gathers use default (RT) policy: table is L2-resident.
    for (int idx = tid; idx < MTILE * AKP; idx += NTHREADS) {
        int row = idx / AKP;
        int d   = idx - row * AKP;
        float v = 0.0f;
        if (d < EMB) {
            int cnt = sCnt[row];
            const int* rp = sRows[row];
            float s = 0.0f;
            #pragma unroll 4
            for (int c = 0; c < 4; ++c) {
                if (c < cnt) s += emb[rp[c] * EMB + d];
            }
            v = s * sInv[row];
        }
        sA[idx] = (_Float16)v;
    }
    __syncthreads();

    // ---- WMMA GEMM: A(LDS, 64xKP) x Wt^T (global, KPx768) ------------------
    // Wave w computes ALL 4 M-tiles for N-tiles [w*6, w*6+6): each B fragment
    // load is amortized over 4 WMMAs (4x less L2 traffic than 1 tile/wave).
    const int lane = tid & 31;
    const int wave = tid >> 5;
    const int nlan = lane & 15;
    const int kf   = (lane < 16) ? 0 : 8;   // 16-bit A/B 16x32 lane K layout
    const int mofs = (lane < 16) ? 0 : 8;   // C/D layout: VGPR r -> M=r / M=r+8

    const _Float16* ap = &sA[nlan * AKP + kf];   // + mt*16*AKP + k*32 per frag

    for (int j = 0; j < 6; ++j) {
        const int hcol = (wave * 6 + j) * 16 + nlan;
        const _Float16* bp = Wt + (size_t)hcol * KP + kf;

        // prefetch next N-tile's B panel into L2/L0 while we crunch this one
        if (j < 5)
            __builtin_prefetch((const void*)(bp + (size_t)16 * KP), 0, 3);

        v8f c0 = {}, c1 = {}, c2 = {}, c3 = {};
        #pragma unroll
        for (int k = 0; k < KSTEPS; ++k) {
            FragU bf;
            bf.u[0] = *(const uint4*)(bp + k * 32);
            bf.u[1] = *(const uint4*)(bp + k * 32 + 16);

            const _Float16* apk = ap + k * 32;
            FragU a0, a1, a2, a3;
            a0.u[0] = *(const uint4*)(apk);
            a0.u[1] = *(const uint4*)(apk + 16);
            a1.u[0] = *(const uint4*)(apk + 16 * AKP);
            a1.u[1] = *(const uint4*)(apk + 16 * AKP + 16);
            a2.u[0] = *(const uint4*)(apk + 32 * AKP);
            a2.u[1] = *(const uint4*)(apk + 32 * AKP + 16);
            a3.u[0] = *(const uint4*)(apk + 48 * AKP);
            a3.u[1] = *(const uint4*)(apk + 48 * AKP + 16);

            c0 = __builtin_amdgcn_wmma_f32_16x16x32_f16(false, a0.v, false, bf.v, (short)0, c0, false, false);
            c1 = __builtin_amdgcn_wmma_f32_16x16x32_f16(false, a1.v, false, bf.v, (short)0, c1, false, false);
            c2 = __builtin_amdgcn_wmma_f32_16x16x32_f16(false, a2.v, false, bf.v, (short)0, c2, false, false);
            c3 = __builtin_amdgcn_wmma_f32_16x16x32_f16(false, a3.v, false, bf.v, (short)0, c3, false, false);
        }

        const float bv = bias[hcol];
        #pragma unroll
        for (int r = 0; r < 8; ++r) {
            size_t o0 = (size_t)(tok0 +  0 + mofs + r) * HID + hcol;
            size_t o1 = (size_t)(tok0 + 16 + mofs + r) * HID + hcol;
            size_t o2 = (size_t)(tok0 + 32 + mofs + r) * HID + hcol;
            size_t o3 = (size_t)(tok0 + 48 + mofs + r) * HID + hcol;
            // single-touch streams: non-temporal so emb_table/Wt stay in L2
            float x0 = __builtin_nontemporal_load(&xin[o0]);
            float x1 = __builtin_nontemporal_load(&xin[o1]);
            float x2 = __builtin_nontemporal_load(&xin[o2]);
            float x3 = __builtin_nontemporal_load(&xin[o3]);
            __builtin_nontemporal_store(c0[r] + bv + x0, &out[o0]);
            __builtin_nontemporal_store(c1[r] + bv + x1, &out[o1]);
            __builtin_nontemporal_store(c2[r] + bv + x2, &out[o2]);
            __builtin_nontemporal_store(c3[r] + bv + x3, &out[o3]);
        }
    }
}

// ---------------------------------------------------------------------------
extern "C" void kernel_launch(void* const* d_in, const int* in_sizes, int n_in,
                              void* d_out, int out_size, void* d_ws, size_t ws_size,
                              hipStream_t stream) {
    const int*   ids       = (const int*)  d_in[0];   // input_ids     [B*S]
    const float* xin       = (const float*)d_in[1];   // input_tensors [B*S*768]
    const int*   wn_map    = (const int*)  d_in[2];   // [VOCAB*4]
    const int*   wn_counts = (const int*)  d_in[3];   // [VOCAB]
    const float* emb       = (const float*)d_in[4];   // [N_ENT*300]
    const float* W         = (const float*)d_in[5];   // [300*768]
    const float* bias      = (const float*)d_in[6];   // [768]
    float*       out       = (float*)d_out;
    _Float16*    Wt        = (_Float16*)d_ws;         // 768*320*2 = 480 KB

    const int ntok = in_sizes[0];                     // 65536, multiple of MTILE

    wt_build_kernel<<<(HID * KP + NTHREADS - 1) / NTHREADS, NTHREADS, 0, stream>>>(W, Wt);
    wordnet_fused_kernel<<<ntok / MTILE, NTHREADS, 0, stream>>>(
        ids, xin, wn_map, wn_counts, emb, Wt, bias, out);
}
